// RegionProposalNetwork_51041391346207
// MI455X (gfx1250) — matrix operand order repaired and checked by reading, not compiled, not run on hardware
//
#include <hip/hip_runtime.h>
#include <hip/hip_bf16.h>
#include <math.h>

#define AS1 __attribute__((address_space(1)))
#define AS3 __attribute__((address_space(3)))

typedef __attribute__((ext_vector_type(16))) __bf16 v16bf;
typedef __attribute__((ext_vector_type(8)))  float  v8f;
typedef __attribute__((ext_vector_type(4)))  int    v4i;

#define HH    50
#define WWID  63
#define NPOS  3150          // 50*63
#define CIN   512
#define COUT  512
#define KTOT  4608          // 512*9
#define NBATCH 4
#define NA    35            // 7 ratios * 5 scales
#define NANCH 110250        // 3150*35
#define PRE   6000
#define POST  300
#define SORTN 131072        // next pow2 >= 110250

union Frag { uint4 q[2]; v16bf v; };

__device__ __forceinline__ unsigned short f2bf(float f) {
  unsigned int u = __float_as_uint(f);
  u += 0x7FFFu + ((u >> 16) & 1u);
  return (unsigned short)(u >> 16);
}
__device__ __forceinline__ float bf2f(unsigned short h) {
  return __uint_as_float(((unsigned int)h) << 16);
}

__device__ __forceinline__ void async_cp16(const void* g, void* l) {
#if defined(__gfx1250__) && __has_builtin(__builtin_amdgcn_global_load_async_to_lds_b128)
  __builtin_amdgcn_global_load_async_to_lds_b128(
      (AS1 v4i*)g, (AS3 v4i*)l, 0, 0);
#else
  *(uint4*)l = *(const uint4*)g;
#endif
}
__device__ __forceinline__ void wait_async0() {
#if defined(__gfx1250__) && __has_builtin(__builtin_amdgcn_s_wait_asynccnt)
  __builtin_amdgcn_s_wait_asynccnt(0);
#elif defined(__gfx1250__)
  asm volatile("s_wait_asynccnt 0" ::: "memory");
#endif
}

// ---------------------------------------------------------------- converts
__global__ void f32_to_bf16_kernel(const float* __restrict__ src,
                                   unsigned short* __restrict__ dst, int n) {
  int i = blockIdx.x * 256 + threadIdx.x;
  if (i < n) dst[i] = f2bf(src[i]);
}

// ---------------------------------------------------------------- anchors
__global__ void anchor_kernel(float* __restrict__ anch,
                              const int* __restrict__ ph,
                              const int* __restrict__ pw) {
  int i = blockIdx.x * 256 + threadIdx.x;
  if (i >= NANCH) return;
  const float ratios[7] = {0.5f, 0.66f, 0.75f, 1.f, 1.33f, 1.5f, 2.f};
  const float scales[5] = {2.f, 4.f, 8.f, 16.f, 32.f};
  int pos = i / NA, a = i - pos * NA;
  int ri = a / 5, si = a - ri * 5;
  float r = ratios[ri], s = scales[si];
  float h = 16.f * s * sqrtf(r);
  float w = 16.f * s * sqrtf(1.f / r);
  int y = pos / WWID, x = pos - y * WWID;
  float sy = (float)y * 16.f, sx = (float)x * 16.f;
  float y1 = 8.f - 0.5f * h + sy, x1 = 8.f - 0.5f * w + sx;
  float y2 = 8.f + 0.5f * h + sy, x2 = 8.f + 0.5f * w + sx;
  float Hc = (float)(ph[0] - 1), Wc = (float)(pw[0] - 1);
  anch[i * 4 + 0] = fminf(fmaxf(y1, 0.f), Hc);
  anch[i * 4 + 1] = fminf(fmaxf(x1, 0.f), Wc);
  anch[i * 4 + 2] = fminf(fmaxf(y2, 0.f), Hc);
  anch[i * 4 + 3] = fminf(fmaxf(x2, 0.f), Wc);
}

// ---------------------------------------------------------------- conv3x3 (WMMA implicit GEMM)
__device__ __forceinline__ void stageA(unsigned char* dst,
                                       const unsigned short* __restrict__ Wb,
                                       int m0, int k0, int tid) {
  // 128 rows x 32 K bf16 (64B payload per row), LDS pitch 80B; 512 x 16B chunks
  for (int i = tid; i < 512; i += 256) {
    int row = i >> 2, cq = i & 3;
    const unsigned char* g =
        (const unsigned char*)(Wb + (size_t)(m0 + row) * KTOT + k0) + cq * 16;
    async_cp16(g, dst + row * 80 + cq * 16);
  }
}

__device__ __forceinline__ void stageB(unsigned char* dst,
                                       const unsigned short* __restrict__ Fb,
                                       int n, int p0, int k0, int tid) {
  // on-the-fly im2col: 128 positions x 16 k-pairs (dwords), pitch 80B
  for (int idx = tid; idx < 2048; idx += 256) {
    int pl = idx >> 4, kp = idx & 15;
    int pos = p0 + pl;
    unsigned int pack = 0u;
    if (pos < NPOS) {
      int y = pos / WWID, x = pos - y * WWID;
#pragma unroll
      for (int e = 0; e < 2; ++e) {
        int k = k0 + kp * 2 + e;
        int c = k / 9, q = k - c * 9;
        int r = q / 3, s = q - r * 3;
        int iy = y + r - 1, ix = x + s - 1;
        unsigned int hv = 0u;
        if (iy >= 0 && iy < HH && ix >= 0 && ix < WWID)
          hv = Fb[((size_t)n * CIN + c) * NPOS + iy * WWID + ix];
        pack |= hv << (16 * e);
      }
    }
    *(unsigned int*)(dst + pl * 80 + kp * 4) = pack;
  }
}

__global__ __launch_bounds__(256) void conv3x3_wmma(
    const unsigned short* __restrict__ Wb,  // [512][4608] bf16
    const unsigned short* __restrict__ Fb,  // [4][512][3150] bf16
    const float* __restrict__ bias,
    unsigned short* __restrict__ Hb)        // [4][512][3150] bf16 (post-ReLU)
{
  __shared__ __align__(16) unsigned char smA[2][128 * 80];
  __shared__ __align__(16) unsigned char smB[2][128 * 80];

  const int tid = threadIdx.x;
  const int n  = blockIdx.z;
  const int m0 = blockIdx.x * 128;
  const int p0 = blockIdx.y * 128;

  const int wave = tid >> 5, lane = tid & 31;
  const int wm = wave >> 2, wn = wave & 3;   // 2x4 wave grid -> 64M x 32N per wave
  const int lm = lane & 15, lh = lane >> 4;

  v8f acc[4][2] = {};

  const int KS = KTOT / 32;  // 144
  int cur = 0;
  stageA(smA[cur], Wb, m0, 0, tid);
  stageB(smB[cur], Fb, n, p0, 0, tid);

  for (int ks = 0; ks < KS; ++ks) {
    wait_async0();
    __syncthreads();
    if (ks + 1 < KS) {
      stageA(smA[cur ^ 1], Wb, m0, (ks + 1) * 32, tid);
      stageB(smB[cur ^ 1], Fb, n, p0, (ks + 1) * 32, tid);
    }
    const unsigned char* A = smA[cur];
    const unsigned char* B = smB[cur];

    Frag af[4];
#pragma unroll
    for (int tm = 0; tm < 4; ++tm) {
      int base = (wm * 64 + tm * 16 + lm) * 80 + lh * 16;
      af[tm].q[0] = *(const uint4*)(A + base);
      af[tm].q[1] = *(const uint4*)(A + base + 32);
    }
    Frag bq[2];
#pragma unroll
    for (int tn = 0; tn < 2; ++tn) {
      int base = (wn * 32 + tn * 16 + lm) * 80 + lh * 16;
      bq[tn].q[0] = *(const uint4*)(B + base);
      bq[tn].q[1] = *(const uint4*)(B + base + 32);
    }
#pragma unroll
    for (int tm = 0; tm < 4; ++tm)
#pragma unroll
      for (int tn = 0; tn < 2; ++tn)
        acc[tm][tn] = __builtin_amdgcn_wmma_f32_16x16x32_bf16(
            false, af[tm].v, false, bq[tn].v, (short)0, acc[tm][tn], false, false);

    __syncthreads();
    cur ^= 1;
  }

  // epilogue: bias + ReLU, store bf16 [n][oc][pos]
#pragma unroll
  for (int tm = 0; tm < 4; ++tm) {
    int oc_base = m0 + wm * 64 + tm * 16 + 8 * lh;
#pragma unroll
    for (int tn = 0; tn < 2; ++tn) {
      int pos = p0 + wn * 32 + tn * 16 + lm;
      if (pos < NPOS) {
#pragma unroll
        for (int r = 0; r < 8; ++r) {
          int oc = oc_base + r;
          float v = acc[tm][tn][r] + bias[oc];
          v = v > 0.f ? v : 0.f;
          Hb[((size_t)n * COUT + oc) * NPOS + pos] = f2bf(v);
        }
      }
    }
  }
}

// ---------------------------------------------------------------- 1x1 heads
__global__ void heads_score(const unsigned short* __restrict__ Hb,
                            const float* __restrict__ sw,
                            const float* __restrict__ sb,
                            float* __restrict__ out_scores,
                            float* __restrict__ fgs) {
  int n = blockIdx.z, a = blockIdx.y;
  int p = blockIdx.x * 128 + threadIdx.x;
  if (p >= NPOS) return;
  float s0 = sb[2 * a], s1 = sb[2 * a + 1];
  const unsigned short* hp = Hb + (size_t)n * COUT * NPOS + p;
  const float* w0 = sw + (size_t)(2 * a) * COUT;
  const float* w1 = w0 + COUT;
  for (int c = 0; c < COUT; ++c) {
    float hv = bf2f(hp[(size_t)c * NPOS]);
    s0 = fmaf(hv, w0[c], s0);
    s1 = fmaf(hv, w1[c], s1);
  }
  size_t o = (size_t)n * NANCH + (size_t)p * NA + a;
  out_scores[o * 2 + 0] = s0;
  out_scores[o * 2 + 1] = s1;
  float mx = fmaxf(s0, s1);
  float e0 = __expf(s0 - mx), e1 = __expf(s1 - mx);
  fgs[o] = e1 / (e0 + e1);
}

__global__ void heads_loc(const unsigned short* __restrict__ Hb,
                          const float* __restrict__ lw,
                          const float* __restrict__ lb,
                          float* __restrict__ out_locs) {
  int n = blockIdx.z, oc = blockIdx.y;  // 0..139
  int p = blockIdx.x * 128 + threadIdx.x;
  if (p >= NPOS) return;
  float s = lb[oc];
  const unsigned short* hp = Hb + (size_t)n * COUT * NPOS + p;
  const float* w = lw + (size_t)oc * COUT;
  for (int c = 0; c < COUT; ++c)
    s = fmaf(bf2f(hp[(size_t)c * NPOS]), w[c], s);
  int a = oc >> 2, j = oc & 3;
  out_locs[((size_t)n * NANCH + (size_t)p * NA + a) * 4 + j] = s;
}

// ---------------------------------------------------------------- decode + sort keys
__device__ __forceinline__ unsigned int score_key(float f) {
  unsigned int u = __float_as_uint(f);
  unsigned int m = (u & 0x80000000u) ? ~u : (u | 0x80000000u);  // ascending float
  return ~m;                                                    // ascending key = descending score
}

__global__ void decode_kernel(const float* __restrict__ anch,
                              const float* __restrict__ locs,
                              const float* __restrict__ fgs,
                              const int* __restrict__ ph,
                              const int* __restrict__ pw,
                              float* __restrict__ boxes,
                              unsigned int* __restrict__ keys,
                              unsigned int* __restrict__ vals) {
  int i = blockIdx.x * 256 + threadIdx.x;
  int n = blockIdx.y;
  if (i >= SORTN) return;
  if (i >= NANCH) {
    keys[(size_t)n * SORTN + i] = 0xFFFFFFFFu;
    vals[(size_t)n * SORTN + i] = 0u;
    return;
  }
  const float* A = anch + (size_t)i * 4;
  const float* L = locs + ((size_t)n * NANCH + i) * 4;
  float ah = A[2] - A[0], aw = A[3] - A[1];
  float cy = A[0] + 0.5f * ah, cx = A[1] + 0.5f * aw;
  float ncy = L[0] * ah + cy, ncx = L[1] * aw + cx;
  float nh = __expf(L[2]) * ah, nw = __expf(L[3]) * aw;
  float Hf = (float)ph[0], Wf = (float)pw[0];
  float y1 = fminf(fmaxf(ncy - 0.5f * nh, 0.f), Hf);
  float x1 = fminf(fmaxf(ncx - 0.5f * nw, 0.f), Wf);
  float y2 = fminf(fmaxf(ncy + 0.5f * nh, 0.f), Hf);
  float x2 = fminf(fmaxf(ncx + 0.5f * nw, 0.f), Wf);
  float* b = boxes + ((size_t)n * NANCH + i) * 4;
  b[0] = y1; b[1] = x1; b[2] = y2; b[3] = x2;
  bool ok = ((y2 - y1) >= 16.f) && ((x2 - x1) >= 16.f);
  float sc = ok ? fgs[(size_t)n * NANCH + i] : -INFINITY;
  keys[(size_t)n * SORTN + i] = score_key(sc);
  vals[(size_t)n * SORTN + i] = (unsigned int)i;
}

// ---------------------------------------------------------------- bitonic sort step
__global__ void bitonic_step(unsigned int* __restrict__ keys,
                             unsigned int* __restrict__ vals,
                             unsigned int j, unsigned int k) {
  size_t base = (size_t)blockIdx.y * SORTN;
  unsigned int i = blockIdx.x * 256 + threadIdx.x;
  unsigned int ixj = i ^ j;
  if (ixj > i) {
    unsigned int ka = keys[base + i], kb = keys[base + ixj];
    bool up = ((i & k) == 0);
    if (up ? (ka > kb) : (ka < kb)) {
      keys[base + i] = kb; keys[base + ixj] = ka;
      unsigned int va = vals[base + i];
      vals[base + i] = vals[base + ixj]; vals[base + ixj] = va;
    }
  }
}

// ---------------------------------------------------------------- gather top-PRE + validity
__global__ void gather_top(const unsigned int* __restrict__ keys,
                           const unsigned int* __restrict__ vals,
                           const float* __restrict__ boxes,
                           float* __restrict__ topb,
                           int* __restrict__ keep) {
  int i = blockIdx.x * 256 + threadIdx.x;
  if (i >= NBATCH * PRE) return;
  int n = i / PRE, r = i - n * PRE;
  unsigned int key = keys[(size_t)n * SORTN + r];
  unsigned int idx = vals[(size_t)n * SORTN + r];
  unsigned int m = ~key;
  unsigned int u = (m & 0x80000000u) ? (m ^ 0x80000000u) : ~m;
  float sc = __uint_as_float(u);
  int valid = (key != 0xFFFFFFFFu) && (sc > -3.0e38f);
  const float* b = boxes + ((size_t)n * NANCH + idx) * 4;
  float* t = topb + ((size_t)n * PRE + r) * 4;
  t[0] = b[0]; t[1] = b[1]; t[2] = b[2]; t[3] = b[3];
  keep[(size_t)n * PRE + r] = valid;
}

// ---------------------------------------------------------------- greedy NMS (one block / image)
__global__ __launch_bounds__(256) void nms_kernel(const float* __restrict__ topb,
                                                  int* __restrict__ keep) {
  __shared__ unsigned char kf[PRE];
  int n = blockIdx.x, tid = threadIdx.x;
  const float* tb = topb + (size_t)n * PRE * 4;
  int* kg = keep + (size_t)n * PRE;
  for (int r = tid; r < PRE; r += 256) kf[r] = (unsigned char)kg[r];
  __syncthreads();
  for (int i = 0; i < PRE; ++i) {
    if (kf[i]) {
      float y1 = tb[i * 4 + 0], x1 = tb[i * 4 + 1];
      float y2 = tb[i * 4 + 2], x2 = tb[i * 4 + 3];
      float ai = fmaxf(y2 - y1, 0.f) * fmaxf(x2 - x1, 0.f);
      for (int jj = i + 1 + tid; jj < PRE; jj += 256) {
        if (!kf[jj]) continue;
        float by1 = tb[jj * 4 + 0], bx1 = tb[jj * 4 + 1];
        float by2 = tb[jj * 4 + 2], bx2 = tb[jj * 4 + 3];
        float aj = fmaxf(by2 - by1, 0.f) * fmaxf(bx2 - bx1, 0.f);
        float ty = fmaxf(y1, by1), tx = fmaxf(x1, bx1);
        float by = fminf(y2, by2), bx = fminf(x2, bx2);
        float inter = fmaxf(by - ty, 0.f) * fmaxf(bx - tx, 0.f);
        float iou = inter / fmaxf(ai + aj - inter, 1e-9f);
        if (iou > 0.7f) kf[jj] = 0;
      }
    }
    __syncthreads();
  }
  for (int r = tid; r < PRE; r += 256) kg[r] = (int)kf[r];
}

// ---------------------------------------------------------------- select POST rois
__global__ void select_kernel(const float* __restrict__ topb,
                              const int* __restrict__ keep,
                              float* __restrict__ rois,
                              float* __restrict__ ridx) {
  int n = blockIdx.x;
  if (threadIdx.x != 0) return;
  const float* tb = topb + (size_t)n * PRE * 4;
  const int* kp = keep + (size_t)n * PRE;
  int cnt = 0;
  for (int r = 0; r < PRE && cnt < POST; ++r) {
    if (kp[r]) {
      for (int c = 0; c < 4; ++c)
        rois[((size_t)n * POST + cnt) * 4 + c] = tb[r * 4 + c];
      ++cnt;
    }
  }
  for (; cnt < POST; ++cnt)
    for (int c = 0; c < 4; ++c) rois[((size_t)n * POST + cnt) * 4 + c] = 0.f;
  for (int r = 0; r < POST; ++r) ridx[n * POST + r] = (float)n;
}

// ================================================================ launch
extern "C" void kernel_launch(void* const* d_in, const int* in_sizes, int n_in,
                              void* d_out, int out_size, void* d_ws, size_t ws_size,
                              hipStream_t stream) {
  const float* features = (const float*)d_in[0];  // [4,512,50,63]
  const float* conv_w   = (const float*)d_in[1];  // [512,512,3,3] == [512][4608]
  const float* conv_b   = (const float*)d_in[2];
  const float* score_w  = (const float*)d_in[3];  // [70,512]
  const float* score_b  = (const float*)d_in[4];
  const float* loc_w    = (const float*)d_in[5];  // [140,512]
  const float* loc_b    = (const float*)d_in[6];
  const int*   img_h    = (const int*)d_in[7];
  const int*   img_w    = (const int*)d_in[8];
  (void)in_sizes; (void)n_in; (void)out_size; (void)ws_size;

  float* out = (float*)d_out;
  float* o_locs   = out;                                   // 4*110250*4
  float* o_scores = o_locs + (size_t)NBATCH * NANCH * 4;   // 4*110250*2
  float* o_rois   = o_scores + (size_t)NBATCH * NANCH * 2; // 4*300*4
  float* o_ridx   = o_rois + (size_t)NBATCH * POST * 4;    // 4*300
  float* o_anch   = o_ridx + (size_t)NBATCH * POST;        // 110250*4

  // workspace carve
  unsigned char* w = (unsigned char*)d_ws;
  size_t off = 0;
  auto carve = [&](size_t bytes) {
    unsigned char* p = w + off;
    off += (bytes + 255) & ~(size_t)255;
    return p;
  };
  const size_t FEAT_E = (size_t)NBATCH * CIN * NPOS;       // 6,451,200
  const size_t WB_E   = (size_t)COUT * KTOT;               // 2,359,296
  unsigned short* featb = (unsigned short*)carve(FEAT_E * 2);
  unsigned short* wb    = (unsigned short*)carve(WB_E * 2);
  unsigned short* hb    = (unsigned short*)carve(FEAT_E * 2);
  float*        fgs     = (float*)carve((size_t)NBATCH * NANCH * 4);
  float*        boxes   = (float*)carve((size_t)NBATCH * NANCH * 4 * 4);
  unsigned int* keys    = (unsigned int*)carve((size_t)NBATCH * SORTN * 4);
  unsigned int* vals    = (unsigned int*)carve((size_t)NBATCH * SORTN * 4);
  float*        topb    = (float*)carve((size_t)NBATCH * PRE * 4 * 4);
  int*          keep    = (int*)carve((size_t)NBATCH * PRE * 4);

  // 1) f32 -> bf16 copies (features + conv weights)
  f32_to_bf16_kernel<<<dim3((unsigned)((FEAT_E + 255) / 256)), 256, 0, stream>>>(
      features, featb, (int)FEAT_E);
  f32_to_bf16_kernel<<<dim3((unsigned)((WB_E + 255) / 256)), 256, 0, stream>>>(
      conv_w, wb, (int)WB_E);

  // 2) anchors (clipped) -> output (also consumed by decode)
  anchor_kernel<<<dim3((NANCH + 255) / 256), 256, 0, stream>>>(o_anch, img_h, img_w);

  // 3) 3x3 conv as WMMA implicit GEMM
  conv3x3_wmma<<<dim3(COUT / 128, (NPOS + 127) / 128, NBATCH), 256, 0, stream>>>(
      wb, featb, conv_b, hb);

  // 4) 1x1 heads + softmax fg
  heads_score<<<dim3((NPOS + 127) / 128, NA, NBATCH), 128, 0, stream>>>(
      hb, score_w, score_b, o_scores, fgs);
  heads_loc<<<dim3((NPOS + 127) / 128, 4 * NA, NBATCH), 128, 0, stream>>>(
      hb, loc_w, loc_b, o_locs);

  // 5) decode boxes + sort keys (with -inf padding to SORTN)
  decode_kernel<<<dim3(SORTN / 256, NBATCH), 256, 0, stream>>>(
      o_anch, o_locs, fgs, img_h, img_w, boxes, keys, vals);

  // 6) bitonic sort (descending score), per image
  for (unsigned int k = 2; k <= (unsigned int)SORTN; k <<= 1)
    for (unsigned int j = k >> 1; j > 0; j >>= 1)
      bitonic_step<<<dim3(SORTN / 256, NBATCH), 256, 0, stream>>>(keys, vals, j, k);

  // 7) gather top-PRE boxes + validity
  gather_top<<<dim3((NBATCH * PRE + 255) / 256), 256, 0, stream>>>(
      keys, vals, boxes, topb, keep);

  // 8) greedy NMS
  nms_kernel<<<dim3(NBATCH), 256, 0, stream>>>(topb, keep);

  // 9) select first POST kept -> rois + indices
  select_kernel<<<dim3(NBATCH), 32, 0, stream>>>(topb, keep, o_rois, o_ridx);
}
